// Seq2Seq_66640712565052
// MI455X (gfx1250) — matrix operand order repaired and checked by reading, not compiled, not run on hardware
//
#include <hip/hip_runtime.h>
#include <hip/hip_bf16.h>
#include <math.h>

// Problem constants (from reference)
#define B_   64
#define S_   128
#define T_   64
#define V_   32000
#define E_   512
#define H_   512
#define L_   2
#define BOS_ 1
#define GH   (4 * H_)     // 2048 gate width
#define KC   (E_ + H_)    // 1024 concatenated K

typedef __bf16 bf16_t;
typedef __bf16 v16bf __attribute__((ext_vector_type(16)));
typedef __bf16 v8bf  __attribute__((ext_vector_type(8)));
typedef float  v8f   __attribute__((ext_vector_type(8)));
typedef unsigned int u32x4 __attribute__((ext_vector_type(4)));
typedef int          i32x8 __attribute__((ext_vector_type(8)));
typedef int          i32x4 __attribute__((ext_vector_type(4)));

__device__ __forceinline__ float sigmoidf_(float x) { return 1.0f / (1.0f + expf(-x)); }

// ---------------------------------------------------------------------------
// TDM: stage a rows x cols (bf16) panel from global into LDS (2-D descriptor).
// D# layout per CDNA5 ISA:
//   group0: [1:0]=count=1, [63:32]=lds_addr, [120:64]=global_addr, [127:126]=type=2
//   group1: [17:16]=data_size(1 => 2 bytes), [79:48]=tensor_dim0, [111:80]=tensor_dim1,
//           [127:112]=tile_dim0, [143:128]=tile_dim1, [207:160]=tensor_dim0_stride
// Toolchain here is the 6-arg builtin variant (clang-23 / therock-10.0):
//   (u32x4 g0, i32x8 g1, i32x4 g2, i32x4 g3, i32x8 g4, i32 cpol)
// ---------------------------------------------------------------------------
__device__ __forceinline__ void tdm_load_panel(const bf16_t* gsrc, unsigned lds_off,
                                               int rows, int cols) {
    unsigned long long ga = (unsigned long long)(size_t)gsrc;
    u32x4 g0;
    g0[0] = 1u;                                    // count=1 (valid), user mode
    g0[1] = lds_off;                               // lds_addr (bytes)
    g0[2] = (unsigned)(ga & 0xffffffffu);          // global_addr[31:0]
    g0[3] = (unsigned)((ga >> 32) & 0x01ffffffu)   // global_addr[56:32]
          | (2u << 30);                            // type = 2 ("image")
    i32x8 g1;
    g1[0] = 1 << 16;                               // data_size=1 (2B), wg_mask=0
    g1[1] = (cols & 0xffff) << 16;                 // tensor_dim0[15:0]
    g1[2] = (int)(((unsigned)cols >> 16) | (((unsigned)rows & 0xffffu) << 16));
    g1[3] = (int)((((unsigned)rows) >> 16) | (((unsigned)cols & 0xffffu) << 16)); // tile_dim0 = cols
    g1[4] = rows & 0xffff;                         // tile_dim1 = rows, tile_dim2 = 0
    g1[5] = cols;                                  // tensor_dim0_stride[31:0]
    g1[6] = 0;
    g1[7] = 0;
    i32x4 z4 = {0, 0, 0, 0};
    i32x8 z8 = {0, 0, 0, 0, 0, 0, 0, 0};
    __builtin_amdgcn_tensor_load_to_lds(g0, g1, z4, z4, z8, 0);
}

// ---------------------------------------------------------------------------
// Setup: build concatenated [Wih | Whh] weights in bf16 (N=4H rows, K=E+H)
// ---------------------------------------------------------------------------
__global__ void build_wcat_kernel(const float* __restrict__ Wih,
                                  const float* __restrict__ Whh,
                                  bf16_t* __restrict__ out, int total) {
    int idx = blockIdx.x * blockDim.x + threadIdx.x;
    if (idx >= total) return;
    int k   = idx & (KC - 1);
    int rem = idx >> 10;          // l*GH + n
    int n   = rem & (GH - 1);
    int l   = rem >> 11;
    float v = (k < E_) ? Wih[((size_t)l * GH + n) * E_ + k]
                       : Whh[((size_t)l * GH + n) * H_ + (k - E_)];
    out[idx] = (bf16_t)v;
}

__global__ void cvt_f32_bf16_kernel(const float* __restrict__ in,
                                    bf16_t* __restrict__ out, int n) {
    int idx = blockIdx.x * blockDim.x + threadIdx.x;
    if (idx < n) out[idx] = (bf16_t)in[idx];
}

__global__ void init_state_kernel(float* __restrict__ h, float* __restrict__ c,
                                  bf16_t* __restrict__ hbf, int* __restrict__ inp) {
    int idx = blockIdx.x * blockDim.x + threadIdx.x;
    if (idx < L_ * B_ * H_) {
        h[idx] = 0.0f; c[idx] = 0.0f; hbf[idx] = (bf16_t)0.0f;
    }
    if (idx < B_) inp[idx] = BOS_;
}

// ---------------------------------------------------------------------------
// Prep: pack [embedding(token) | h_bf16] as bf16, 64 x 1024
// ---------------------------------------------------------------------------
__global__ void prep_embed_kernel(const float* __restrict__ emb,
                                  const int* __restrict__ tokens, int tstride, int toff,
                                  const bf16_t* __restrict__ hbf,
                                  bf16_t* __restrict__ inpcat) {
    int idx = blockIdx.x * blockDim.x + threadIdx.x;   // B_*KC
    if (idx >= B_ * KC) return;
    int b = idx >> 10, k = idx & (KC - 1);
    bf16_t v;
    if (k < E_) {
        int tok = tokens[b * tstride + toff];
        v = (bf16_t)emb[(size_t)tok * E_ + k];
    } else {
        v = hbf[b * H_ + (k - E_)];
    }
    inpcat[idx] = v;
}

__global__ void prep_hidden_kernel(const bf16_t* __restrict__ hbf0,
                                   const bf16_t* __restrict__ hbf1,
                                   bf16_t* __restrict__ inpcat) {
    int idx = blockIdx.x * blockDim.x + threadIdx.x;
    if (idx >= B_ * KC) return;
    int b = idx >> 10, k = idx & (KC - 1);
    inpcat[idx] = (k < E_) ? hbf0[b * H_ + k] : hbf1[b * H_ + (k - E_)];
}

// ---------------------------------------------------------------------------
// WMMA GEMM: C[M,N](f32) = A[M,K](bf16) @ W[N,K](bf16)^T
// TDM stages the shared A panel into LDS once per block (all 8 waves reuse it);
// each wave owns a 16x64 output strip = 4 accumulators, so one LDS A-fragment
// feeds 4 v_wmma_f32_16x16x32_bf16 per K-step. B streams from L2 with
// immediate-offset b128 loads (N-tiles are 16*K*2 bytes apart).
// ---------------------------------------------------------------------------
__global__ void gemm_bf16_wmma_kernel(const bf16_t* __restrict__ A,
                                      const bf16_t* __restrict__ W,
                                      float* __restrict__ C,
                                      int M, int N, int K) {
    extern __shared__ __align__(16) char lds_raw[];
    bf16_t* As = (bf16_t*)lds_raw;

    // wave 0 issues the TDM panel load, waits TENSORcnt, then block barrier
    if ((threadIdx.x >> 5) == 0) {
        tdm_load_panel(A, 0u, M, K);
        __builtin_amdgcn_s_wait_tensorcnt(0);
    }
    __syncthreads();

    const int wavesPerBlock = blockDim.x >> 5;
    const int wave    = blockIdx.x * wavesPerBlock + (threadIdx.x >> 5);
    const int tilesM  = M >> 4;
    const int tilesNG = N >> 6;                 // groups of 4 N-tiles
    if (wave >= tilesM * tilesNG) return;       // wave-uniform (after barrier)
    const int tm   = wave % tilesM;
    const int n0   = (wave / tilesM) << 6;
    const int lane = threadIdx.x & 31;
    const int lo   = lane & 15;
    const int hi   = lane >> 4;

    // A fragment from LDS (ISA 16-bit A layout: lanes 0-15 K0-7/16-23, 16-31 K8-15/24-31)
    const bf16_t* arow = As + (size_t)(tm * 16 + lo) * K + hi * 8;
    // B fragment: lane = output column, K contiguous (row of W)
    const bf16_t* brow = W + (size_t)(n0 + lo) * K + hi * 8;
    const size_t  bstride = (size_t)16 * K;     // elements between N-tiles

    v8f acc[4];
#pragma unroll
    for (int j = 0; j < 4; ++j) acc[j] = (v8f){0.f, 0.f, 0.f, 0.f, 0.f, 0.f, 0.f, 0.f};

    for (int k0 = 0; k0 < K; k0 += 32) {
        if (k0 + 256 < K) __builtin_prefetch(brow + k0 + 256, 0, 1);
        v8bf a0 = *(const v8bf*)(arow + k0);
        v8bf a1 = *(const v8bf*)(arow + k0 + 16);
        v16bf av;
#pragma unroll
        for (int i = 0; i < 8; ++i) { av[i] = a0[i]; av[8 + i] = a1[i]; }
#pragma unroll
        for (int j = 0; j < 4; ++j) {
            const bf16_t* bp = brow + (size_t)j * bstride + k0;
            v8bf b0 = *(const v8bf*)(bp);
            v8bf b1 = *(const v8bf*)(bp + 16);
            v16bf bv;
#pragma unroll
            for (int i = 0; i < 8; ++i) { bv[i] = b0[i]; bv[8 + i] = b1[i]; }
            acc[j] = __builtin_amdgcn_wmma_f32_16x16x32_bf16(
                false, av, false, bv, (short)0, acc[j], false, false);
        }
    }

    // C/D layout: VGPR i -> row (hi*8 + i), col = lo
#pragma unroll
    for (int j = 0; j < 4; ++j) {
        float* crow = C + (size_t)(tm * 16 + hi * 8) * N + (n0 + j * 16 + lo);
#pragma unroll
        for (int i = 0; i < 8; ++i) crow[(size_t)i * N] = acc[j][i];
    }
}

// ---------------------------------------------------------------------------
// LSTM cell elementwise: gates(f32) + bias -> (h,c) update with optional mask
// ---------------------------------------------------------------------------
__global__ void lstm_cell_kernel(const float* __restrict__ gates,
                                 const float* __restrict__ bias,
                                 float* __restrict__ h, float* __restrict__ c,
                                 bf16_t* __restrict__ hbf,
                                 const int* __restrict__ x_len, int t, int use_mask) {
    int idx = blockIdx.x * blockDim.x + threadIdx.x;   // B_*H_
    if (idx >= B_ * H_) return;
    int b = idx / H_, j = idx % H_;
    const float* g = gates + (size_t)b * GH;
    float gi = g[j]          + bias[j];
    float gf = g[H_ + j]     + bias[H_ + j];
    float gg = g[2 * H_ + j] + bias[2 * H_ + j];
    float go = g[3 * H_ + j] + bias[3 * H_ + j];
    float c_old = c[idx], h_old = h[idx];
    float c2 = sigmoidf_(gf) * c_old + sigmoidf_(gi) * tanhf(gg);
    float h2 = sigmoidf_(go) * tanhf(c2);
    if (use_mask && !(t < x_len[b])) { c2 = c_old; h2 = h_old; }
    c[idx] = c2; h[idx] = h2; hbf[idx] = (bf16_t)h2;
}

// ---------------------------------------------------------------------------
// Generator epilogue: per batch row, log-softmax gather + argmax + token select
// ---------------------------------------------------------------------------
__global__ void gen_select_kernel(const float* __restrict__ logits,
                                  const int* __restrict__ y_seqs,
                                  const int* __restrict__ teacher, int t,
                                  int* __restrict__ inp, float* __restrict__ out) {
    const int b = blockIdx.x;
    const float* row = logits + (size_t)b * V_;
    __shared__ float smax[256];
    __shared__ int   sidx[256];
    __shared__ float ssum[256];
    const int tid = threadIdx.x;

    float m = -INFINITY; int mi = 0;
    for (int j = tid; j < V_; j += 256) {
        float v = row[j];
        if (v > m) { m = v; mi = j; }
    }
    smax[tid] = m; sidx[tid] = mi;
    __syncthreads();
    for (int s = 128; s > 0; s >>= 1) {
        if (tid < s) {
            float v2 = smax[tid + s]; int i2 = sidx[tid + s];
            if (v2 > smax[tid] || (v2 == smax[tid] && i2 < sidx[tid])) {
                smax[tid] = v2; sidx[tid] = i2;
            }
        }
        __syncthreads();
    }
    const float gmax = smax[0];
    const int   gidx = sidx[0];
    __syncthreads();

    float s = 0.0f;
    for (int j = tid; j < V_; j += 256) s += expf(row[j] - gmax);
    ssum[tid] = s;
    __syncthreads();
    for (int st = 128; st > 0; st >>= 1) {
        if (tid < st) ssum[tid] += ssum[tid + st];
        __syncthreads();
    }
    if (tid == 0) {
        float logZ = gmax + logf(ssum[0]);
        int y = y_seqs[b * T_ + (t + 1)];
        out[(size_t)b * (T_ - 1) + t] = row[y] - logZ;
        inp[b] = (teacher[t] > 0) ? y : gidx;
    }
}

// ---------------------------------------------------------------------------
extern "C" void kernel_launch(void* const* d_in, const int* in_sizes, int n_in,
                              void* d_out, int out_size, void* d_ws, size_t ws_size,
                              hipStream_t stream) {
    const int*   x_seqs  = (const int*)  d_in[0];
    const int*   x_len   = (const int*)  d_in[1];
    const int*   y_seqs  = (const int*)  d_in[2];
    const int*   teacher = (const int*)  d_in[3];
    const float* src_emb = (const float*)d_in[4];
    const float* tgt_emb = (const float*)d_in[5];
    const float* enc_Wih = (const float*)d_in[6];
    const float* enc_Whh = (const float*)d_in[7];
    const float* enc_b   = (const float*)d_in[8];
    const float* dec_Wih = (const float*)d_in[9];
    const float* dec_Whh = (const float*)d_in[10];
    const float* dec_b   = (const float*)d_in[11];
    const float* gen_W   = (const float*)d_in[12];
    float* outp = (float*)d_out;

    // Workspace layout
    char* ws = (char*)d_ws;
    size_t off = 0;
    auto alloc = [&](size_t bytes) -> char* {
        char* p = ws + off;
        off = (off + bytes + 255) & ~(size_t)255;
        return p;
    };
    bf16_t* WcatE  = (bf16_t*)alloc((size_t)L_ * GH * KC * 2);   // 8.4 MB
    bf16_t* WcatD  = (bf16_t*)alloc((size_t)L_ * GH * KC * 2);   // 8.4 MB
    bf16_t* genWb  = (bf16_t*)alloc((size_t)V_ * H_ * 2);        // 32.8 MB (L2-resident)
    float*  hstate = (float*) alloc((size_t)L_ * B_ * H_ * 4);
    float*  cstate = (float*) alloc((size_t)L_ * B_ * H_ * 4);
    bf16_t* hbf    = (bf16_t*)alloc((size_t)L_ * B_ * H_ * 2);
    bf16_t* inpcat = (bf16_t*)alloc((size_t)B_ * KC * 2);
    float*  gates  = (float*) alloc((size_t)B_ * GH * 4);
    float*  logits = (float*) alloc((size_t)B_ * V_ * 4);
    int*    inp    = (int*)   alloc((size_t)B_ * 4);
    (void)ws_size; (void)n_in; (void)in_sizes; (void)out_size;

    // ---- one-time (per launch) weight conversion + state init ----
    {
        int total = L_ * GH * KC;
        build_wcat_kernel<<<(total + 255) / 256, 256, 0, stream>>>(enc_Wih, enc_Whh, WcatE, total);
        build_wcat_kernel<<<(total + 255) / 256, 256, 0, stream>>>(dec_Wih, dec_Whh, WcatD, total);
        int gtot = V_ * H_;
        cvt_f32_bf16_kernel<<<(gtot + 255) / 256, 256, 0, stream>>>(gen_W, genWb, gtot);
        init_state_kernel<<<(L_ * B_ * H_ + 255) / 256, 256, 0, stream>>>(hstate, cstate, hbf, inp);
    }

    const int prepGrid = (B_ * KC + 255) / 256;              // 256 blocks
    const int cellGrid = (B_ * H_ + 255) / 256;              // 128 blocks
    const int gateWaves   = (B_ / 16) * (GH / 64);           // 128 waves (16x64 per wave)
    const int gateBlocks  = (gateWaves + 7) / 8;             // 16 blocks
    const size_t gateShm  = (size_t)B_ * KC * 2;             // 128 KB A panel
    const int logitWaves  = (B_ / 16) * (V_ / 64);           // 2000 waves
    const int logitBlocks = (logitWaves + 7) / 8;            // 250 blocks
    const size_t logitShm = (size_t)B_ * H_ * 2;             // 64 KB A panel

    // ---- encoder: 128 timesteps, 2 layers, length-masked ----
    for (int t = 0; t < S_; ++t) {
        prep_embed_kernel<<<prepGrid, 256, 0, stream>>>(src_emb, x_seqs, S_, t,
                                                        hbf, inpcat);
        gemm_bf16_wmma_kernel<<<gateBlocks, 256, gateShm, stream>>>(inpcat, WcatE, gates,
                                                                    B_, GH, KC);
        lstm_cell_kernel<<<cellGrid, 256, 0, stream>>>(gates, enc_b,
                                                       hstate, cstate, hbf,
                                                       x_len, t, 1);
        prep_hidden_kernel<<<prepGrid, 256, 0, stream>>>(hbf, hbf + B_ * H_, inpcat);
        gemm_bf16_wmma_kernel<<<gateBlocks, 256, gateShm, stream>>>(inpcat, WcatE + (size_t)GH * KC,
                                                                    gates, B_, GH, KC);
        lstm_cell_kernel<<<cellGrid, 256, 0, stream>>>(gates, enc_b + GH,
                                                       hstate + B_ * H_, cstate + B_ * H_,
                                                       hbf + B_ * H_, x_len, t, 1);
    }

    // ---- decoder: 63 timesteps; states carry over from encoder finals ----
    for (int t = 0; t < T_ - 1; ++t) {
        prep_embed_kernel<<<prepGrid, 256, 0, stream>>>(tgt_emb, inp, 1, 0,
                                                        hbf, inpcat);
        gemm_bf16_wmma_kernel<<<gateBlocks, 256, gateShm, stream>>>(inpcat, WcatD, gates,
                                                                    B_, GH, KC);
        lstm_cell_kernel<<<cellGrid, 256, 0, stream>>>(gates, dec_b,
                                                       hstate, cstate, hbf,
                                                       x_len, t, 0);
        prep_hidden_kernel<<<prepGrid, 256, 0, stream>>>(hbf, hbf + B_ * H_, inpcat);
        gemm_bf16_wmma_kernel<<<gateBlocks, 256, gateShm, stream>>>(inpcat, WcatD + (size_t)GH * KC,
                                                                    gates, B_, GH, KC);
        lstm_cell_kernel<<<cellGrid, 256, 0, stream>>>(gates, dec_b + GH,
                                                       hstate + B_ * H_, cstate + B_ * H_,
                                                       hbf + B_ * H_, x_len, t, 0);
        // generator: 64x512 @ 512x32000 (dominant GEMM; genWb bf16 lives in L2)
        gemm_bf16_wmma_kernel<<<logitBlocks, 256, logitShm, stream>>>(hbf + B_ * H_, genWb,
                                                                      logits, B_, V_, H_);
        gen_select_kernel<<<B_, 256, 0, stream>>>(logits, y_seqs, teacher, t, inp, outp);
    }
}
